// LinearAttention_43052752175481
// MI455X (gfx1250) — matrix operand order repaired and checked by reading, not compiled, not run on hardware
//
#include <hip/hip_runtime.h>
#include <hip/hip_bf16.h>
#include <cstdint>

// ---------------------------------------------------------------------------
// Linear attention (RoPE + elu+1 feature map) for MI455X / gfx1250.
// GEMMs: V_WMMA_F32_16X16X32_F16 (f16 in, f32 acc) -> memory-bound pipeline.
// B-operands pre-transposed so K is contiguous, staged to LDS with
// GLOBAL_LOAD_ASYNC_TO_LDS_B128 (+ s_wait_asynccnt), double-buffered in k_kv.
// ---------------------------------------------------------------------------

typedef _Float16 v16h __attribute__((ext_vector_type(16)));
typedef _Float16 v8h  __attribute__((ext_vector_type(8)));
typedef _Float16 v4h  __attribute__((ext_vector_type(4)));
typedef float    v8f  __attribute__((ext_vector_type(8)));

#define BQ     4
#define SEQ    8192
#define DMODEL 512
#define NH     8
#define DH     64
#define INNER  512          // NH*DH
#define QKVN   1536         // 3*INNER
#define ROWS_TOT (BQ*SEQ)   // 32768

union V16U { v16h v; v8h h[2]; };

// A-frag (16-bit, 16x32 MxK): lane m=lane&15; lane<16 K {kk..+7, kk+16..+23},
// lane>=16 K {kk+8..+15, kk+24..+31}. All offsets 16B-aligned.
__device__ __forceinline__ v16h load_a_frag(const _Float16* rowbase, int kk, int khalf){
  V16U u;
  u.h[0] = *(const v8h*)(rowbase + kk + khalf*8);
  u.h[1] = *(const v8h*)(rowbase + kk + 16 + khalf*8);
  return u.v;
}
// B-frag (16-bit, 32x16 KxN) from K-contiguous (transposed) storage.
__device__ __forceinline__ v16h load_b_frag(const _Float16* colbase, int kk, int khalf){
  V16U u;
  const _Float16* p = colbase + kk + khalf*8;
  u.h[0] = *(const v8h*)(p);
  u.h[1] = *(const v8h*)(p + 16);
  return u.v;
}
__device__ __forceinline__ v8f wmma16(v16h a, v16h b, v8f c){
  return __builtin_amdgcn_wmma_f32_16x16x32_f16(false, a, false, b, (short)0, c, false, false);
}
// Async 16B global->LDS copy (CDNA5 TDM-lite path, ASYNCcnt-tracked).
__device__ __forceinline__ void async_b128(void* lds, const void* gaddr){
  unsigned lofs = (unsigned)(uintptr_t)lds;   // low 32 bits of flat LDS addr = LDS offset
  asm volatile("global_load_async_to_lds_b128 %0, %1, off"
               :: "v"(lofs), "v"(gaddr) : "memory");
}

// ---------------------------------------------------------------------------
// Transpose-convert: d[c*K + k] = (f16) s[k*C + c]
__global__ void k_cvt_t(const float* __restrict__ s, _Float16* __restrict__ d, int K, int C){
  int i = blockIdx.x*256 + threadIdx.x;
  if (i < K*C){
    int k = i / C, c = i - k*C;
    d[(size_t)c*K + k] = (_Float16)s[i];
  }
}
__global__ void k_zero(float* __restrict__ p, int n){
  int i = blockIdx.x*256 + threadIdx.x;
  if (i < n) p[i] = 0.f;
}

// ---------------------------------------------------------------------------
// Kernel 1: qkv = x @ W_qkv + b; RoPE + elu+1 on q/k. Tile 64x64 (one head),
// K=512. 8 waves; wave w: rows (w&3)*16, cols (w>>2)*32. W tile staged via
// async b128 from pre-transposed WqT[c][k]. Outputs:
//   qf  [bh][n][d]  (K-contiguous rows for k_num A-frags)
//   kft [bh][d][n], vft [bh][d][n]  (K(=n)-contiguous rows for k_kv async)
// ---------------------------------------------------------------------------
__global__ __launch_bounds__(256) void k_qkv(const float* __restrict__ x,
                                             const _Float16* __restrict__ WqT,
                                             const float* __restrict__ bq,
                                             _Float16* __restrict__ qf,
                                             _Float16* __restrict__ kft,
                                             _Float16* __restrict__ vft)
{
  __shared__ _Float16 xs[64*80];   // [row][k], stride 160B
  __shared__ _Float16 wt[64*80];   // [col][k]
  __shared__ float    outs[64*68]; // f32 stash for cross-column RoPE pairing
  const int tid = threadIdx.x, lane = tid & 31, wv = tid >> 5;
  const int m0 = blockIdx.x * 64;
  const int c0 = blockIdx.y * 64;
  const int r0 = (wv & 3) * 16;
  const int cc = (wv >> 2) * 32;
  v8f acc0 = {}; v8f acc1 = {};

  for (int k0 = 0; k0 < DMODEL; k0 += 64){
    // W tile: 64 cols x 64 k = 512 x 16B async copies, 2 per thread
    #pragma unroll
    for (int it = 0; it < 2; ++it){
      int o = it*256 + tid;
      int c = o >> 3, c8 = (o & 7) * 8;
      async_b128(&wt[c*80 + c8], WqT + (size_t)(c0 + c)*DMODEL + k0 + c8);
    }
    // x tile: f32 -> f16, float4 loads, v4h LDS stores, fixed trip count
    #pragma unroll
    for (int it = 0; it < 4; ++it){
      int i = it*256 + tid;
      int r = i >> 4, k4 = (i & 15) * 4;
      const float4 xv = *(const float4*)&x[(size_t)(m0 + r)*DMODEL + k0 + k4];
      v4h hv = { (_Float16)xv.x, (_Float16)xv.y, (_Float16)xv.z, (_Float16)xv.w };
      *(v4h*)&xs[r*80 + k4] = hv;
      if (k0 + 64 < DMODEL)
        __builtin_prefetch(&x[(size_t)(m0 + r)*DMODEL + k0 + 64 + k4], 0, 0);
    }
    asm volatile("s_wait_asynccnt 0x0" ::: "memory");
    __syncthreads();
    const _Float16* arow  = xs + (r0 + (lane & 15))*80;
    const _Float16* bcol0 = wt + (cc + (lane & 15))*80;
    const _Float16* bcol1 = wt + (cc + 16 + (lane & 15))*80;
    #pragma unroll
    for (int kk = 0; kk < 64; kk += 32){
      v16h a  = load_a_frag(arow, kk, lane >> 4);
      v16h b0 = load_b_frag(bcol0, kk, lane >> 4);
      v16h b1 = load_b_frag(bcol1, kk, lane >> 4);
      acc0 = wmma16(a, b0, acc0);
      acc1 = wmma16(a, b1, acc1);
    }
    __syncthreads();
  }

  // bias + f32 stash (cols j and j^32 live in different 16-wide tiles)
  #pragma unroll
  for (int r = 0; r < 8; r++){
    int row = r0 + r + 8*(lane >> 4);
    int col = cc + (lane & 15);
    outs[row*68 + col]      = acc0[r] + bq[c0 + col];
    outs[row*68 + col + 16] = acc1[r] + bq[c0 + col + 16];
  }
  __syncthreads();

  const int section = c0 / INNER;          // 0=q, 1=k, 2=v
  const int head    = (c0 % INNER) / DH;
  if (section == 0){
    #pragma unroll
    for (int it = 0; it < 16; ++it){       // consecutive tid -> consecutive j
      int i = it*256 + tid;
      int r = i >> 6, j = i & 63;
      int mg = m0 + r, b = mg >> 13, t = mg & (SEQ - 1);
      float val = outs[r*68 + j];
      int jm = j & 31;
      float invf = __expf(-(float)jm * 0.28782313662f);   // ln(10000)/32
      float sn, cs; __sincosf((float)t * invf, &sn, &cs);
      float other = outs[r*68 + ((j + 32) & 63)];
      float rot = (j < 32) ? (val*cs - other*sn) : (val*cs + other*sn);
      float res = (rot > 0.f) ? (rot + 1.f) : __expf(rot); // elu(x)+1
      qf[(((size_t)(b*NH + head))*SEQ + t)*DH + j] = (_Float16)res;
    }
  } else {
    #pragma unroll
    for (int it = 0; it < 16; ++it){       // consecutive tid -> consecutive t
      int i = it*256 + tid;
      int r = i & 63, j = i >> 6;
      int mg = m0 + r, b = mg >> 13, t = mg & (SEQ - 1);
      float val = outs[r*68 + j];
      float res;
      if (section == 1){
        int jm = j & 31;
        float invf = __expf(-(float)jm * 0.28782313662f);
        float sn, cs; __sincosf((float)t * invf, &sn, &cs);
        float other = outs[r*68 + ((j + 32) & 63)];
        float rot = (j < 32) ? (val*cs - other*sn) : (val*cs + other*sn);
        res = (rot > 0.f) ? (rot + 1.f) : __expf(rot);
      } else {
        res = val;
      }
      size_t dst = (((size_t)(b*NH + head))*DH + j)*SEQ + t;
      if (section == 1) kft[dst] = (_Float16)res; else vft[dst] = (_Float16)res;
    }
  }
}

// ---------------------------------------------------------------------------
// Kernel 2: kv[bh,d,e] = sum_n k[n,d]*v[n,e] (64x64, K=8192 in 8 slices,
// f32 atomics) + ksum. 16 waves, one 16x16 tile each. Operands arrive
// [d][n]/[e][n] so each chunk is 512 async b128 copies (1/thread),
// double-buffered: issue chunk i+1, s_wait_asynccnt 1, compute chunk i.
// ksum is reduced from the staged LDS tiles (no second global sweep).
// ---------------------------------------------------------------------------
__global__ __launch_bounds__(512) void k_kv(const _Float16* __restrict__ kft,
                                            const _Float16* __restrict__ vft,
                                            float* __restrict__ kv,
                                            float* __restrict__ ksum)
{
  __shared__ _Float16 kt[2][64*48];  // [buf][d][n], stride 96B
  __shared__ _Float16 vt[2][64*48];
  __shared__ float    red[64*8];
  const int tid = threadIdx.x, lane = tid & 31, wv = tid >> 5;
  const int bh = blockIdx.y;
  const int nbase0 = blockIdx.x * 1024;
  const int d0 = (wv >> 2) * 16;
  const int e0 = (wv & 3) * 16;
  v8f acc = {};

  // per-thread async stage slot: tid<256 -> kt, else vt; one b128 per chunk
  const int sd = (tid & 255) >> 2;       // 0..63 row (d or e)
  const int sc = (tid & 3) * 8;          // 0,8,16,24 halves within chunk
  const bool isK = tid < 256;
  const _Float16* gsrc = (isK ? kft : vft) + ((size_t)bh*DH + sd)*SEQ + nbase0 + sc;
  const int ldsoff = sd*48 + sc;
  const int kd = tid & 63, kg = tid >> 6; // ksum: thread owns (d, n-subgroup)
  float ks = 0.f;

  async_b128((isK ? kt[0] : vt[0]) + ldsoff, gsrc);   // prime chunk 0
  for (int ch = 0; ch < 1024; ch += 32){
    int buf = (ch >> 5) & 1;
    if (ch + 32 < 1024){
      async_b128((isK ? kt[buf^1] : vt[buf^1]) + ldsoff, gsrc + (ch + 32));
      asm volatile("s_wait_asynccnt 0x1" ::: "memory");
    } else {
      asm volatile("s_wait_asynccnt 0x0" ::: "memory");
    }
    __syncthreads();
    v16h a = load_a_frag(kt[buf] + (d0 + (lane & 15))*48, 0, lane >> 4);
    v16h b = load_b_frag(vt[buf] + (e0 + (lane & 15))*48, 0, lane >> 4);
    acc = wmma16(a, b, acc);
    const _Float16* kr = kt[buf] + kd*48 + kg*4;
    ks += (float)kr[0] + (float)kr[1] + (float)kr[2] + (float)kr[3];
    __syncthreads();
  }
  #pragma unroll
  for (int r = 0; r < 8; r++){
    int dd = d0 + r + 8*(lane >> 4);
    int ee = e0 + (lane & 15);
    atomicAdd(&kv[((size_t)bh*DH + dd)*DH + ee], acc[r]);
  }
  red[kd*8 + kg] = ks;
  __syncthreads();
  if (tid < 64){
    float s = 0.f;
    #pragma unroll
    for (int g = 0; g < 8; g++) s += red[tid*8 + g];
    atomicAdd(&ksum[bh*DH + tid], s);
  }
}

// ---------------------------------------------------------------------------
// Kernel 3: num = qf @ kv (M=8192,N=64,K=64); norm = qf . ksum;
// attn = num/(norm+1e-8) -> f16 [b*n, inner]. A-frags direct from global.
// ---------------------------------------------------------------------------
__global__ __launch_bounds__(256) void k_num(const _Float16* __restrict__ qf,
                                             const float* __restrict__ kv,
                                             const float* __restrict__ ksum,
                                             _Float16* __restrict__ attn)
{
  __shared__ _Float16 kvs[64*80];  // [e][d]
  __shared__ float    kss[64];
  __shared__ float    normv[8*16];
  const int tid = threadIdx.x, lane = tid & 31, wv = tid >> 5;
  const int bh = blockIdx.y;
  const int b = bh >> 3, h = bh & 7;

  #pragma unroll
  for (int it = 0; it < 16; ++it){
    int i = it*256 + tid;
    int d = i >> 6, e = i & 63;
    kvs[e*80 + d] = (_Float16)kv[((size_t)bh*DH + d)*DH + e];
  }
  if (tid < 64) kss[tid] = ksum[bh*DH + tid];
  __syncthreads();

  const int rowbase = blockIdx.x*128 + wv*16;
  const _Float16* arow = qf + ((size_t)bh*SEQ + rowbase + (lane & 15))*DH;
  v16h a0 = load_a_frag(arow,  0, lane >> 4);
  v16h a1 = load_a_frag(arow, 32, lane >> 4);
  v8f acc[4] = {};
  #pragma unroll
  for (int c = 0; c < 4; c++){
    const _Float16* bcol = kvs + (c*16 + (lane & 15))*80;
    v16h b0 = load_b_frag(bcol,  0, lane >> 4);
    v16h b1 = load_b_frag(bcol, 32, lane >> 4);
    acc[c] = wmma16(a0, b0, acc[c]);
    acc[c] = wmma16(a1, b1, acc[c]);
  }
  if (lane < 16){
    const v8h* qr = (const v8h*)(qf + ((size_t)bh*SEQ + rowbase + lane)*DH);
    float s = 0.f;
    #pragma unroll
    for (int c8 = 0; c8 < 8; c8++){
      v8h q8 = qr[c8];
      #pragma unroll
      for (int j = 0; j < 8; j++) s += (float)q8[j] * kss[c8*8 + j];
    }
    normv[wv*16 + lane] = s;
  }
  __syncthreads();
  #pragma unroll
  for (int c = 0; c < 4; c++){
    #pragma unroll
    for (int r = 0; r < 8; r++){
      int ri  = r + 8*(lane >> 4);
      int col = c*16 + (lane & 15);
      float o = acc[c][r] / (normv[wv*16 + ri] + 1e-8f);
      attn[((size_t)b*SEQ + rowbase + ri)*INNER + h*DH + col] = (_Float16)o;
    }
  }
}

// ---------------------------------------------------------------------------
// Kernel 4: out = attn @ W_out + b_out (f32). W tile async-staged from
// pre-transposed WoT[c][k]; A-frags direct from global attn.
// ---------------------------------------------------------------------------
__global__ __launch_bounds__(256) void k_out(const _Float16* __restrict__ attn,
                                             const _Float16* __restrict__ WoT,
                                             const float* __restrict__ bo,
                                             float* __restrict__ out)
{
  __shared__ _Float16 wt[64*80];
  const int tid = threadIdx.x, lane = tid & 31, wv = tid >> 5;
  const int m0 = blockIdx.x*64, c0 = blockIdx.y*64;
  const int r0 = (wv & 3)*16, cc = (wv >> 2)*32;
  v8f acc0 = {}, acc1 = {};

  for (int k0 = 0; k0 < INNER; k0 += 64){
    #pragma unroll
    for (int it = 0; it < 2; ++it){
      int o = it*256 + tid;
      int c = o >> 3, c8 = (o & 7) * 8;
      async_b128(&wt[c*80 + c8], WoT + (size_t)(c0 + c)*INNER + k0 + c8);
    }
    asm volatile("s_wait_asynccnt 0x0" ::: "memory");
    __syncthreads();
    const _Float16* arow  = attn + (size_t)(m0 + r0 + (lane & 15))*INNER + k0;
    const _Float16* bcol0 = wt + (cc + (lane & 15))*80;
    const _Float16* bcol1 = wt + (cc + 16 + (lane & 15))*80;
    #pragma unroll
    for (int kk = 0; kk < 64; kk += 32){
      v16h a  = load_a_frag(arow, kk, lane >> 4);
      v16h b0 = load_b_frag(bcol0, kk, lane >> 4);
      v16h b1 = load_b_frag(bcol1, kk, lane >> 4);
      acc0 = wmma16(a, b0, acc0);
      acc1 = wmma16(a, b1, acc1);
    }
    __syncthreads();
  }
  #pragma unroll
  for (int r = 0; r < 8; r++){
    int row = m0 + r0 + r + 8*(lane >> 4);
    int col = cc + (lane & 15);
    out[(size_t)row*DMODEL + c0 + col]      = acc0[r] + bo[c0 + col];
    out[(size_t)row*DMODEL + c0 + col + 16] = acc1[r] + bo[c0 + col + 16];
  }
}

// ---------------------------------------------------------------------------
extern "C" void kernel_launch(void* const* d_in, const int* in_sizes, int n_in,
                              void* d_out, int out_size, void* d_ws, size_t ws_size,
                              hipStream_t stream)
{
  const float* x    = (const float*)d_in[0];
  const float* Wqkv = (const float*)d_in[1];
  const float* bqkv = (const float*)d_in[2];
  const float* Wout = (const float*)d_in[3];
  const float* bout = (const float*)d_in[4];
  float* out = (float*)d_out;
  (void)in_sizes; (void)n_in; (void)out_size; (void)ws_size;

  char* ws = (char*)d_ws;
  size_t off = 0;
  auto alloc = [&](size_t bytes){ void* p = ws + off; off += (bytes + 255) & ~size_t(255); return p; };
  _Float16* WqT  = (_Float16*)alloc((size_t)DMODEL*QKVN*2);       // [c][k] 1.5 MB
  _Float16* WoT  = (_Float16*)alloc((size_t)INNER*DMODEL*2);      // [c][k] 0.5 MB
  _Float16* qf   = (_Float16*)alloc((size_t)ROWS_TOT*INNER*2);    // [bh][n][d] 32 MB
  _Float16* kft  = (_Float16*)alloc((size_t)ROWS_TOT*INNER*2);    // [bh][d][n] 32 MB
  _Float16* vft  = (_Float16*)alloc((size_t)ROWS_TOT*INNER*2);    // [bh][d][n] 32 MB
  _Float16* attn = (_Float16*)alloc((size_t)ROWS_TOT*INNER*2);    // 32 MB
  float* kv   = (float*)alloc((size_t)BQ*NH*DH*DH*4);             // 512 KB
  float* ksum = (float*)alloc((size_t)BQ*NH*DH*4);                // contiguous after kv

  const int nWq = DMODEL*QKVN, nWo = INNER*DMODEL;
  k_cvt_t<<<(nWq+255)/256, 256, 0, stream>>>(Wqkv, WqT, DMODEL, QKVN);
  k_cvt_t<<<(nWo+255)/256, 256, 0, stream>>>(Wout, WoT, INNER, DMODEL);
  const int nz = BQ*NH*DH*DH + BQ*NH*DH;
  k_zero<<<(nz+255)/256, 256, 0, stream>>>(kv, nz);

  k_qkv<<<dim3(ROWS_TOT/64, QKVN/64), 256, 0, stream>>>(x, WqT, bqkv, qf, kft, vft);
  k_kv <<<dim3(SEQ/1024, BQ*NH),      512, 0, stream>>>(kft, vft, kv, ksum);
  k_num<<<dim3(SEQ/128, BQ*NH),       256, 0, stream>>>(qf, kv, ksum, attn);
  k_out<<<dim3(ROWS_TOT/64, DMODEL/64), 256, 0, stream>>>(attn, WoT, bout, out);
}
